// CrossAttention_17085379904166
// MI455X (gfx1250) — matrix operand order repaired and checked by reading, not compiled, not run on hardware
//
#include <hip/hip_runtime.h>
#include <hip/hip_bf16.h>

// ---------------------------------------------------------------------------
// CDNA5 (gfx1250, wave32) cross-attention:
//   qproj  : q @ Wq + bq, RoPE            -> qh   f16 [B][H][SQ][D]
//   kvproj : kv @ Wkv + bkv, RoPE(k)      -> kf16 [B][SKV][D], vT f16 [B][D][SKV]
//   attn   : flash softmax(QK^T/8) V      -> hbuf f16 [B][SQ][H*D]
//   oproj  : hbuf @ Wo + bo               -> out  f32 [B][SQ][OUT]
// All matmul work via v_wmma_f32_16x16x32_f16; GEMMs are double-buffered
// (register-staged global prefetch overlapped with WMMA on the live buffer).
// ---------------------------------------------------------------------------

typedef __attribute__((ext_vector_type(16))) _Float16  v16h;
typedef __attribute__((ext_vector_type(8)))  float     v8f;
typedef __attribute__((ext_vector_type(4)))  unsigned  u32x4;

#define BATCH 4
#define SEQQ  2048
#define SEQKV 2048
#define HEADS 16
#define DHEAD 64
#define KDIM  1024
#define NOUT  1024
#define LP    40          // padded LDS row length in f16 (80B, conflict-free b128)

union HFrag { v16h v; u32x4 q[2]; };

__device__ __forceinline__ v8f wmma16(const HFrag& a, const HFrag& b, v8f c) {
  return __builtin_amdgcn_wmma_f32_16x16x32_f16(false, a.v, false, b.v,
                                                (short)0, c, false, false);
}

__device__ __forceinline__ v8f zero8() {
  v8f z;
#pragma unroll
  for (int i = 0; i < 8; ++i) z[i] = 0.0f;
  return z;
}

// inv_freq(d) = 4096^(-(d&~1)/64) = exp2(-12*(d&~1)/64)
__device__ __forceinline__ float rope_invfreq(int d) {
  return exp2f(-12.0f * (float)(d & ~1) * (1.0f / 64.0f));
}

// 8 WMMAs on the live LDS buffer (wave tile 32x64 = 2x4 of 16x16)
__device__ __forceinline__ void gemm_step(const _Float16* __restrict__ la,
                                          const _Float16* __restrict__ lb,
                                          int wr, int wc, int lane_lo, int lane_hi,
                                          v8f (&c)[2][4]) {
  HFrag a[2], bf[4];
#pragma unroll
  for (int tr = 0; tr < 2; ++tr) {
    const _Float16* ap = &la[(wr * 32 + tr * 16 + lane_lo) * LP + lane_hi * 16];
    a[tr].q[0] = *(const u32x4*)(ap);
    a[tr].q[1] = *(const u32x4*)(ap + 8);
  }
#pragma unroll
  for (int tc = 0; tc < 4; ++tc) {
    const _Float16* bp = &lb[(wc * 64 + tc * 16 + lane_lo) * LP + lane_hi * 16];
    bf[tc].q[0] = *(const u32x4*)(bp);
    bf[tc].q[1] = *(const u32x4*)(bp + 8);
  }
#pragma unroll
  for (int tr = 0; tr < 2; ++tr)
#pragma unroll
    for (int tc = 0; tc < 4; ++tc)
      c[tr][tc] = wmma16(a[tr], bf[tc], c[tr][tc]);
}

// ---------------- Q projection + RoPE --------------------------------------
__global__ __launch_bounds__(256)
void qproj_kernel(const float* __restrict__ A, const float* __restrict__ W,
                  const float* __restrict__ bias, _Float16* __restrict__ qh) {
  __shared__ _Float16 lds_a[2][128 * LP];
  __shared__ _Float16 lds_b[2][128 * LP];
  const int t = threadIdx.x;
  const int w = t >> 5, lane = t & 31;
  const int lane_lo = lane & 15, lane_hi = lane >> 4;
  const int wr = w & 3, wc = w >> 2;
  const int nBase   = blockIdx.x * 128;
  const int rowBase = blockIdx.y * 128;

  // per-thread tile-fill coordinates (loop-invariant)
  const int ar[4] = { (0*256+t)>>3, (1*256+t)>>3, (2*256+t)>>3, (3*256+t)>>3 };
  const int ac    = (t & 7) << 2;
  const int bk[4] = { (0*256+t)>>5, (1*256+t)>>5, (2*256+t)>>5, (3*256+t)>>5 };
  const int bn    = (t & 31) << 2;

  v8f c[2][4];
#pragma unroll
  for (int i = 0; i < 2; ++i)
#pragma unroll
    for (int j = 0; j < 4; ++j) c[i][j] = zero8();

  float4 ra[4], rb[4];
#pragma unroll
  for (int i = 0; i < 4; ++i) {                   // prologue tile k0=0
    ra[i] = *(const float4*)(A + (size_t)(rowBase + ar[i]) * KDIM + ac);
    rb[i] = *(const float4*)(W + (size_t)bk[i] * NOUT + nBase + bn);
  }
#pragma unroll
  for (int i = 0; i < 4; ++i) {
    _Float16* da = &lds_a[0][ar[i] * LP + ac];
    da[0]=(_Float16)ra[i].x; da[1]=(_Float16)ra[i].y;
    da[2]=(_Float16)ra[i].z; da[3]=(_Float16)ra[i].w;
    _Float16* db = &lds_b[0][bn * LP + bk[i]];
    db[0*LP]=(_Float16)rb[i].x; db[1*LP]=(_Float16)rb[i].y;
    db[2*LP]=(_Float16)rb[i].z; db[3*LP]=(_Float16)rb[i].w;
  }
  __syncthreads();

  for (int k0 = 0; k0 < KDIM; k0 += 32) {
    const int buf = (k0 >> 5) & 1;
    const bool more = (k0 + 32) < KDIM;
    if (more) {                                    // prefetch next tile to regs
#pragma unroll
      for (int i = 0; i < 4; ++i) {
        ra[i] = *(const float4*)(A + (size_t)(rowBase + ar[i]) * KDIM + k0 + 32 + ac);
        rb[i] = *(const float4*)(W + (size_t)(k0 + 32 + bk[i]) * NOUT + nBase + bn);
      }
    }
    gemm_step(lds_a[buf], lds_b[buf], wr, wc, lane_lo, lane_hi, c);
    if (more) {                                    // stage into other buffer
#pragma unroll
      for (int i = 0; i < 4; ++i) {
        _Float16* da = &lds_a[buf ^ 1][ar[i] * LP + ac];
        da[0]=(_Float16)ra[i].x; da[1]=(_Float16)ra[i].y;
        da[2]=(_Float16)ra[i].z; da[3]=(_Float16)ra[i].w;
        _Float16* db = &lds_b[buf ^ 1][bn * LP + bk[i]];
        db[0*LP]=(_Float16)rb[i].x; db[1*LP]=(_Float16)rb[i].y;
        db[2*LP]=(_Float16)rb[i].z; db[3*LP]=(_Float16)rb[i].w;
      }
    }
    __syncthreads();
  }

  // epilogue: bias + RoPE + scatter to [b][h][s][d] f16
#pragma unroll
  for (int tr = 0; tr < 2; ++tr) {
#pragma unroll
    for (int tc = 0; tc < 4; ++tc) {
      const int n  = nBase + wc * 64 + tc * 16 + lane_lo;
      const int hh = n >> 6, d = n & 63;
      const float invf = rope_invfreq(d);
      const float bv = bias[n];
#pragma unroll
      for (int r = 0; r < 8; ++r) {
        int grow = rowBase + wr * 32 + tr * 16 + lane_hi * 8 + r;
        int bb = grow >> 11, s = grow & (SEQQ - 1);
        float val = c[tr][tc][r] + bv;
        float other = __shfl_xor(val, 1, 32);      // partner column d^1
        float ang = (float)s * invf;
        float cs = __cosf(ang), sn = __sinf(ang);
        float outv = (d & 1) ? fmaf(val, cs, other * sn)
                             : fmaf(val, cs, -other * sn);
        qh[(((size_t)bb * HEADS + hh) * SEQQ + s) * DHEAD + d] = (_Float16)outv;
      }
    }
  }
}

// ---------------- KV projection: k (RoPE) + v (transposed) -----------------
__global__ __launch_bounds__(256)
void kvproj_kernel(const float* __restrict__ A, const float* __restrict__ W,
                   const float* __restrict__ bias,
                   _Float16* __restrict__ kf, _Float16* __restrict__ vT) {
  __shared__ _Float16 lds_a[2][128 * LP];
  __shared__ _Float16 lds_b[2][128 * LP];
  const int t = threadIdx.x;
  const int w = t >> 5, lane = t & 31;
  const int lane_lo = lane & 15, lane_hi = lane >> 4;
  const int wr = w & 3, wc = w >> 2;
  const int rowBase = blockIdx.x * 128;

  const int ar[4] = { (0*256+t)>>3, (1*256+t)>>3, (2*256+t)>>3, (3*256+t)>>3 };
  const int ac    = (t & 7) << 2;
  const int bk[4] = { (0*256+t)>>5, (1*256+t)>>5, (2*256+t)>>5, (3*256+t)>>5 };
  const int bn    = (t & 31) << 2;

  v8f c[2][4];
#pragma unroll
  for (int i = 0; i < 2; ++i)
#pragma unroll
    for (int j = 0; j < 4; ++j) c[i][j] = zero8();

  float4 ra[4], rb[4];
#pragma unroll
  for (int i = 0; i < 4; ++i) {
    ra[i] = *(const float4*)(A + (size_t)(rowBase + ar[i]) * KDIM + ac);
    rb[i] = *(const float4*)(W + (size_t)bk[i] * 128 + bn);
  }
#pragma unroll
  for (int i = 0; i < 4; ++i) {
    _Float16* da = &lds_a[0][ar[i] * LP + ac];
    da[0]=(_Float16)ra[i].x; da[1]=(_Float16)ra[i].y;
    da[2]=(_Float16)ra[i].z; da[3]=(_Float16)ra[i].w;
    _Float16* db = &lds_b[0][bn * LP + bk[i]];
    db[0*LP]=(_Float16)rb[i].x; db[1*LP]=(_Float16)rb[i].y;
    db[2*LP]=(_Float16)rb[i].z; db[3*LP]=(_Float16)rb[i].w;
  }
  __syncthreads();

  for (int k0 = 0; k0 < KDIM; k0 += 32) {
    const int buf = (k0 >> 5) & 1;
    const bool more = (k0 + 32) < KDIM;
    if (more) {
#pragma unroll
      for (int i = 0; i < 4; ++i) {
        ra[i] = *(const float4*)(A + (size_t)(rowBase + ar[i]) * KDIM + k0 + 32 + ac);
        rb[i] = *(const float4*)(W + (size_t)(k0 + 32 + bk[i]) * 128 + bn);
      }
    }
    gemm_step(lds_a[buf], lds_b[buf], wr, wc, lane_lo, lane_hi, c);
    if (more) {
#pragma unroll
      for (int i = 0; i < 4; ++i) {
        _Float16* da = &lds_a[buf ^ 1][ar[i] * LP + ac];
        da[0]=(_Float16)ra[i].x; da[1]=(_Float16)ra[i].y;
        da[2]=(_Float16)ra[i].z; da[3]=(_Float16)ra[i].w;
        _Float16* db = &lds_b[buf ^ 1][bn * LP + bk[i]];
        db[0*LP]=(_Float16)rb[i].x; db[1*LP]=(_Float16)rb[i].y;
        db[2*LP]=(_Float16)rb[i].z; db[3*LP]=(_Float16)rb[i].w;
      }
    }
    __syncthreads();
  }

#pragma unroll
  for (int tr = 0; tr < 2; ++tr) {
#pragma unroll
    for (int tc = 0; tc < 4; ++tc) {
      const int n = wc * 64 + tc * 16 + lane_lo;
      const float bv = bias[n];
      if (n < 64) {                                // K head: RoPE + [b][s][d]
        const int d = n;
        const float invf = rope_invfreq(d);
#pragma unroll
        for (int r = 0; r < 8; ++r) {
          int grow = rowBase + wr * 32 + tr * 16 + lane_hi * 8 + r;
          int bb = grow >> 11, s = grow & (SEQKV - 1);
          float val = c[tr][tc][r] + bv;
          float other = __shfl_xor(val, 1, 32);
          float ang = (float)s * invf;
          float cs = __cosf(ang), sn = __sinf(ang);
          float outv = (d & 1) ? fmaf(val, cs, other * sn)
                               : fmaf(val, cs, -other * sn);
          kf[((size_t)bb * SEQKV + s) * DHEAD + d] = (_Float16)outv;
        }
      } else {                                     // V head: transpose [b][d][s]
        const int d = n - 64;
#pragma unroll
        for (int r = 0; r < 8; ++r) {
          int grow = rowBase + wr * 32 + tr * 16 + lane_hi * 8 + r;
          int bb = grow >> 11, s = grow & (SEQKV - 1);
          float val = c[tr][tc][r] + bv;
          vT[((size_t)bb * DHEAD + d) * SEQKV + s] = (_Float16)val;
        }
      }
    }
  }
}

// ---------------- Flash attention (one K/V head shared by 16 Q heads) ------
__global__ __launch_bounds__(256)
void attn_kernel(const _Float16* __restrict__ qh, const _Float16* __restrict__ kf,
                 const _Float16* __restrict__ vT, _Float16* __restrict__ hbuf) {
  __shared__ _Float16 lds_p[8 * 16 * LP];          // per-wave 16x32 P patch
  const int t = threadIdx.x;
  const int w = t >> 5, lane = t & 31;
  const int lane_lo = lane & 15, lane_hi = lane >> 4;
  const int blk = blockIdx.x;
  const int qt = blk & 15, hh = (blk >> 4) & 15, bb = blk >> 8;
  const int qrow0 = qt * 128 + w * 16;

  // Q A-fragments for both k=32 d-slabs (held for the whole kv loop)
  HFrag aq[2];
  const _Float16* qp =
      qh + (((size_t)bb * HEADS + hh) * SEQQ + qrow0 + lane_lo) * DHEAD;
#pragma unroll
  for (int cslab = 0; cslab < 2; ++cslab) {
    aq[cslab].q[0] = *(const u32x4*)(qp + cslab * 32 + lane_hi * 16);
    aq[cslab].q[1] = *(const u32x4*)(qp + cslab * 32 + 8 + lane_hi * 16);
  }

  float m[8], l[8];
#pragma unroll
  for (int r = 0; r < 8; ++r) { m[r] = -3.0e38f; l[r] = 0.0f; }
  v8f acc[4];
#pragma unroll
  for (int nt = 0; nt < 4; ++nt) acc[nt] = zero8();

  const _Float16* kbase = kf + (size_t)bb * SEQKV * DHEAD;
  const _Float16* vbase = vT + (size_t)bb * DHEAD * SEQKV;
  _Float16* pw = &lds_p[w * 16 * LP];
  const float scale = 0.125f;                      // 1/sqrt(64)

  for (int kv0 = 0; kv0 < SEQKV; kv0 += 32) {
    // issue ALL K and V fragment loads up front so global latency overlaps
    // with the S WMMAs + softmax VALU below
    HFrag bk[2][2], bv[4];
#pragma unroll
    for (int kt = 0; kt < 2; ++kt) {
      const _Float16* kp =
          kbase + (size_t)(kv0 + kt * 16 + lane_lo) * DHEAD + lane_hi * 16;
      bk[kt][0].q[0] = *(const u32x4*)(kp);
      bk[kt][0].q[1] = *(const u32x4*)(kp + 8);
      bk[kt][1].q[0] = *(const u32x4*)(kp + 32);
      bk[kt][1].q[1] = *(const u32x4*)(kp + 40);
    }
#pragma unroll
    for (int nt = 0; nt < 4; ++nt) {
      const _Float16* vp =
          vbase + (size_t)(nt * 16 + lane_lo) * SEQKV + kv0 + lane_hi * 16;
      bv[nt].q[0] = *(const u32x4*)(vp);
      bv[nt].q[1] = *(const u32x4*)(vp + 8);
    }

    // S = Q K^T for two 16-key tiles
    v8f s01[2];
#pragma unroll
    for (int kt = 0; kt < 2; ++kt) {
      v8f sacc = zero8();
      sacc = wmma16(aq[0], bk[kt][0], sacc);
      sacc = wmma16(aq[1], bk[kt][1], sacc);
      s01[kt] = sacc;
    }

    // online softmax update (rows striped across 16 lanes per half)
    float corr[8];
#pragma unroll
    for (int r = 0; r < 8; ++r) {
      float s0 = s01[0][r] * scale;
      float s1 = s01[1][r] * scale;
      float tm = fmaxf(s0, s1);
#pragma unroll
      for (int off = 1; off < 16; off <<= 1)
        tm = fmaxf(tm, __shfl_xor(tm, off, 32));
      float mn = fmaxf(m[r], tm);
      corr[r] = __expf(m[r] - mn);
      m[r] = mn;
      float p0 = __expf(s0 - mn);
      float p1 = __expf(s1 - mn);
      pw[(lane_hi * 8 + r) * LP + lane_lo]      = (_Float16)p0;
      pw[(lane_hi * 8 + r) * LP + 16 + lane_lo] = (_Float16)p1;
      float rs = p0 + p1;
#pragma unroll
      for (int off = 1; off < 16; off <<= 1)
        rs += __shfl_xor(rs, off, 32);
      l[r] = l[r] * corr[r] + rs;
    }
#pragma unroll
    for (int nt = 0; nt < 4; ++nt)
#pragma unroll
      for (int r = 0; r < 8; ++r) acc[nt][r] *= corr[r];

    // P back as A-fragment (same-wave LDS round trip, no barrier needed)
    HFrag pa;
    const _Float16* pp = pw + lane_lo * LP + lane_hi * 16;
    pa.q[0] = *(const u32x4*)(pp);
    pa.q[1] = *(const u32x4*)(pp + 8);

    // O += P V  (vT gives key-contiguous B-fragments)
#pragma unroll
    for (int nt = 0; nt < 4; ++nt)
      acc[nt] = wmma16(pa, bv[nt], acc[nt]);
  }

  // normalize and write h in [b][s][h*64+d] f16
#pragma unroll
  for (int r = 0; r < 8; ++r) {
    float invl = 1.0f / l[r];
    int row = qrow0 + lane_hi * 8 + r;
#pragma unroll
    for (int nt = 0; nt < 4; ++nt) {
      int col = hh * 64 + nt * 16 + lane_lo;
      hbuf[((size_t)bb * SEQQ + row) * (HEADS * DHEAD) + col] =
          (_Float16)(acc[nt][r] * invl);
    }
  }
}

// ---------------- Output projection (f16 A) --------------------------------
__global__ __launch_bounds__(256)
void oproj_kernel(const _Float16* __restrict__ Ah, const float* __restrict__ W,
                  const float* __restrict__ bias, float* __restrict__ out) {
  __shared__ _Float16 lds_a[2][128 * LP];
  __shared__ _Float16 lds_b[2][128 * LP];
  const int t = threadIdx.x;
  const int w = t >> 5, lane = t & 31;
  const int lane_lo = lane & 15, lane_hi = lane >> 4;
  const int wr = w & 3, wc = w >> 2;
  const int nBase   = blockIdx.x * 128;
  const int rowBase = blockIdx.y * 128;

  const int ar[2] = { (0*256+t)>>2, (1*256+t)>>2 };
  const int ac8   = (t & 3) << 3;
  const int bk[4] = { (0*256+t)>>5, (1*256+t)>>5, (2*256+t)>>5, (3*256+t)>>5 };
  const int bn    = (t & 31) << 2;

  v8f c[2][4];
#pragma unroll
  for (int i = 0; i < 2; ++i)
#pragma unroll
    for (int j = 0; j < 4; ++j) c[i][j] = zero8();

  u32x4 raw[2];
  float4 rb[4];
#pragma unroll
  for (int i = 0; i < 2; ++i)
    raw[i] = *(const u32x4*)(Ah + (size_t)(rowBase + ar[i]) * KDIM + ac8);
#pragma unroll
  for (int i = 0; i < 4; ++i)
    rb[i] = *(const float4*)(W + (size_t)bk[i] * NOUT + nBase + bn);
#pragma unroll
  for (int i = 0; i < 2; ++i)
    *(u32x4*)&lds_a[0][ar[i] * LP + ac8] = raw[i];
#pragma unroll
  for (int i = 0; i < 4; ++i) {
    _Float16* db = &lds_b[0][bn * LP + bk[i]];
    db[0*LP]=(_Float16)rb[i].x; db[1*LP]=(_Float16)rb[i].y;
    db[2*LP]=(_Float16)rb[i].z; db[3*LP]=(_Float16)rb[i].w;
  }
  __syncthreads();

  for (int k0 = 0; k0 < KDIM; k0 += 32) {
    const int buf = (k0 >> 5) & 1;
    const bool more = (k0 + 32) < KDIM;
    if (more) {
#pragma unroll
      for (int i = 0; i < 2; ++i)
        raw[i] = *(const u32x4*)(Ah + (size_t)(rowBase + ar[i]) * KDIM + k0 + 32 + ac8);
#pragma unroll
      for (int i = 0; i < 4; ++i)
        rb[i] = *(const float4*)(W + (size_t)(k0 + 32 + bk[i]) * NOUT + nBase + bn);
    }
    gemm_step(lds_a[buf], lds_b[buf], wr, wc, lane_lo, lane_hi, c);
    if (more) {
#pragma unroll
      for (int i = 0; i < 2; ++i)
        *(u32x4*)&lds_a[buf ^ 1][ar[i] * LP + ac8] = raw[i];
#pragma unroll
      for (int i = 0; i < 4; ++i) {
        _Float16* db = &lds_b[buf ^ 1][bn * LP + bk[i]];
        db[0*LP]=(_Float16)rb[i].x; db[1*LP]=(_Float16)rb[i].y;
        db[2*LP]=(_Float16)rb[i].z; db[3*LP]=(_Float16)rb[i].w;
      }
    }
    __syncthreads();
  }

#pragma unroll
  for (int tr = 0; tr < 2; ++tr) {
#pragma unroll
    for (int tc = 0; tc < 4; ++tc) {
      const int n = nBase + wc * 64 + tc * 16 + lane_lo;
      const float bv = bias[n];
#pragma unroll
      for (int r = 0; r < 8; ++r) {
        int grow = rowBase + wr * 32 + tr * 16 + lane_hi * 8 + r;
        out[(size_t)grow * NOUT + n] = c[tr][tc][r] + bv;
      }
    }
  }
}

// ---------------------------------------------------------------------------
extern "C" void kernel_launch(void* const* d_in, const int* in_sizes, int n_in,
                              void* d_out, int out_size, void* d_ws, size_t ws_size,
                              hipStream_t stream) {
  (void)in_sizes; (void)n_in; (void)out_size; (void)ws_size;
  const float* q   = (const float*)d_in[0];
  const float* kv  = (const float*)d_in[1];
  const float* Wq  = (const float*)d_in[2];
  const float* bq  = (const float*)d_in[3];
  const float* Wkv = (const float*)d_in[4];
  const float* bkv = (const float*)d_in[5];
  const float* Wo  = (const float*)d_in[6];
  const float* bo  = (const float*)d_in[7];

  // workspace carve (f16): qh 16.8MB, k 1MB, vT 1MB, hbuf 16.8MB  (~35.7MB)
  _Float16* qh   = (_Float16*)d_ws;
  _Float16* kf   = qh + (size_t)BATCH * HEADS * SEQQ * DHEAD;
  _Float16* vT   = kf + (size_t)BATCH * SEQKV * DHEAD;
  _Float16* hbuf = vT + (size_t)BATCH * SEQKV * DHEAD;

  const int rowTilesQ  = (BATCH * SEQQ)  / 128;   // 64
  const int rowTilesKV = (BATCH * SEQKV) / 128;   // 64

  qproj_kernel<<<dim3(NOUT / 128, rowTilesQ), 256, 0, stream>>>(q, Wq, bq, qh);
  kvproj_kernel<<<rowTilesKV, 256, 0, stream>>>(kv, Wkv, bkv, kf, vT);
  attn_kernel<<<BATCH * HEADS * (SEQQ / 128), 256, 0, stream>>>(qh, kf, vT, hbuf);
  oproj_kernel<<<dim3(NOUT / 128, rowTilesQ), 256, 0, stream>>>(hbuf, Wo, bo,
                                                                (float*)d_out);
}
